// PlanningModule_49452253446292
// MI455X (gfx1250) — compile-verified, matrix operand order checked by reading, not run on hardware
//
// CDNA5 / gfx1250 implementation of the BitNet PlanningModule.
//
// Key optimizations (reasoned from MI455X roofline, see analysis):
//  * context_memory is quantized to int8 ONCE (it is step-invariant under
//    bitlinear's fake-quant), turning 1.37 TFLOP of per-step K/V projections
//    into ~1 GB of bandwidth-bound int8 passes (~41 us @ 23.3 TB/s).
//  * scores are computed as (gamma_k/s) * (Wk^T q) . xq  (exact algebra in the
//    quantized domain); cv as Wv . (sum_s aw*gamma_v/s * xq_s) + bv.
//  * All dense layers use V_WMMA_I32_16X16X64_IU8: BitNet = int8 act x ternary
//    weight, so integer WMMA with i32 accumulation is the exact precision path.
//  * gru_whh is dead (multiplied by quantized zeros); only gru_bhh survives.
//
#include <hip/hip_runtime.h>
#include <cstdint>

#define B_ 32
#define S_ 2048
#define H_ 1024
#define P_ 1024
#define A_ 1024
#define ST_ 5
#define EPS_ 1e-5f

typedef __attribute__((ext_vector_type(8))) int v8i_t;

// ---------------- wave reductions (wave32) ----------------
__device__ __forceinline__ float wred_f(float v){
#pragma unroll
  for(int o=16;o;o>>=1) v += __shfl_down(v,o,32);
  return v;
}
__device__ __forceinline__ int wred_i(int v){
#pragma unroll
  for(int o=16;o;o>>=1) v += __shfl_down(v,o,32);
  return v;
}

// ---------------- weight quantization: gamma = mean|w|+EPS, wq = clip(round(w/gamma),-1,1)
__global__ void absum_kernel(const float* __restrict__ w, int n, float* __restrict__ part){
  __shared__ float red[256];
  float s = 0.f;
  for(int i = blockIdx.x*blockDim.x + threadIdx.x; i < n; i += gridDim.x*blockDim.x)
    s += fabsf(w[i]);
  red[threadIdx.x] = s; __syncthreads();
  for(int t=128;t;t>>=1){ if(threadIdx.x<t) red[threadIdx.x] += red[threadIdx.x+t]; __syncthreads(); }
  if(threadIdx.x==0) part[blockIdx.x] = red[0];
}

__global__ void fin_gamma_kernel(const float* __restrict__ part, int nb, float invn, float* __restrict__ g){
  __shared__ float red[256];
  float s = 0.f;
  for(int i=threadIdx.x;i<nb;i+=256) s += part[i];
  red[threadIdx.x]=s; __syncthreads();
  for(int t=128;t;t>>=1){ if(threadIdx.x<t) red[threadIdx.x]+=red[threadIdx.x+t]; __syncthreads(); }
  if(threadIdx.x==0) *g = red[0]*invn + EPS_;
}

__global__ void quantw_kernel(const float* __restrict__ w, int n, const float* __restrict__ g,
                              int8_t* __restrict__ wq){
  float ig = 1.f / (*g);
  for(int i = blockIdx.x*blockDim.x + threadIdx.x; i < n; i += gridDim.x*blockDim.x){
    float v = rintf(w[i]*ig);
    v = fminf(fmaxf(v, -1.f), 1.f);
    wq[i] = (int8_t)v;
  }
}

// ---------------- per-token activation quantization: s=127/clip(max|x|,EPS); store recip = 1/s
__global__ void quant_rows_kernel(const float* __restrict__ x, int cols,
                                  int8_t* __restrict__ xq, float* __restrict__ recip){
  int row = blockIdx.x;
  const float* xr = x + (size_t)row*cols;
  __shared__ float red[256];
  float m = 0.f;
  for(int c=threadIdx.x;c<cols;c+=256) m = fmaxf(m, fabsf(xr[c]));
  red[threadIdx.x]=m; __syncthreads();
  for(int t=128;t;t>>=1){ if(threadIdx.x<t) red[threadIdx.x]=fmaxf(red[threadIdx.x],red[threadIdx.x+t]); __syncthreads(); }
  float mx = fmaxf(red[0], EPS_);
  if(threadIdx.x==0) recip[row] = mx * (1.f/127.f);
  float s127 = 127.f / mx;
  int8_t* xo = xq + (size_t)row*cols;
  for(int c=threadIdx.x;c<cols;c+=256){
    float v = rintf(xr[c]*s127);
    v = fminf(fmaxf(v,-128.f),127.f);
    xo[c] = (int8_t)v;
  }
}

// ---------------- int8 GEMM on V_WMMA_I32_16X16X64_IU8 ----------------
// out[M,N] = dequant( Xq[M,K] . Wq[N,K]^T ) ; dequant = acc * gamma * recip[row] + bias
// 4 waves / block, each wave owns one 16x16 tile, K-loop of 16x16x64 WMMAs.
// Fragment layout per ISA 7.12.2 (8-bit A 16x64; B symmetric with N on lanes).
__global__ void gemm_iu8_kernel(const int8_t* __restrict__ Xq, const float* __restrict__ xrecip,
                                const int8_t* __restrict__ Wq, const float* __restrict__ gamma,
                                const float* __restrict__ bias, const float* __restrict__ addsrc,
                                float* __restrict__ out, int N, int K, int act){
  int lane = threadIdx.x & 31;
  int wv   = threadIdx.x >> 5;
  int n0 = (blockIdx.x*4 + wv) * 16;
  int m0 = blockIdx.y * 16;
  int half = lane >> 4;
  int l15  = lane & 15;
  const int8_t* arow = Xq + (size_t)(m0 + l15)*K;
  const int8_t* brow = Wq + (size_t)(n0 + l15)*K;
  v8i_t acc = {};
  for(int kb=0; kb<K; kb+=64){
    v8i_t a, b;
#pragma unroll
    for(int j=0;j<8;++j){
      int k = kb + ((j>>2)<<5) + (((j>>1)&1)<<4) + ((j&1)<<2) + half*8;
      a[j] = *(const int*)(arow + k);
      b[j] = *(const int*)(brow + k);
    }
    acc = __builtin_amdgcn_wmma_i32_16x16x64_iu8(true, a, true, b, acc, false, false);
  }
  float g = *gamma;
  int ncol = n0 + l15;
  float bv = bias ? bias[ncol] : 0.f;
#pragma unroll
  for(int r=0;r<8;++r){
    int mrow = m0 + r + half*8;
    float v = (float)acc[r] * g * xrecip[mrow] + bv;
    if(addsrc) v += addsrc[(size_t)mrow*N + ncol];
    if(act==1) v = tanhf(v);
    out[(size_t)mrow*N + ncol] = v;
  }
}

// ---------------- u[b,h] = sum_a q[b,a] * Wk_int[a,h]   (fp x ternary, tiny) ----------------
__global__ void u_kernel(const float* __restrict__ q, const int8_t* __restrict__ Wkq,
                         float* __restrict__ u){
  int b = blockIdx.x;
  int h0 = threadIdx.x*4;
  float a0=0.f,a1=0.f,a2=0.f,a3=0.f;
  const float* qb = q + (size_t)b*A_;
  for(int a=0;a<A_;++a){
    float qa = qb[a];
    int w4 = *(const int*)(Wkq + (size_t)a*H_ + h0);
    a0 += qa * (float)(int8_t)(w4);
    a1 += qa * (float)(int8_t)(w4>>8);
    a2 += qa * (float)(int8_t)(w4>>16);
    a3 += qa * (float)(int8_t)(w4>>24);
  }
  float* ub = u + (size_t)b*H_ + h0;
  ub[0]=a0; ub[1]=a1; ub[2]=a2; ub[3]=a3;
}

__global__ void qbk_kernel(const float* __restrict__ q, const float* __restrict__ bk,
                           float* __restrict__ qbk){
  int b = blockIdx.x;
  __shared__ float red[256];
  float s = 0.f;
  for(int a=threadIdx.x;a<A_;a+=256) s += q[(size_t)b*A_+a]*bk[a];
  red[threadIdx.x]=s; __syncthreads();
  for(int t=128;t;t>>=1){ if(threadIdx.x<t) red[threadIdx.x]+=red[threadIdx.x+t]; __syncthreads(); }
  if(threadIdx.x==0) qbk[b]=red[0];
}

// ---------------- scores[b,s] = (gamma_k*recip_bs*(u_b . xq_bs) + q.bk) / sqrt(A) ----------------
__global__ void scores_kernel(const int8_t* __restrict__ xq, const float* __restrict__ ctxr,
                              const float* __restrict__ u, const float* __restrict__ qbk,
                              const float* __restrict__ gk, float* __restrict__ sc){
  int gtid = blockIdx.x*blockDim.x + threadIdx.x;
  int wid = gtid >> 5, lane = gtid & 31;
  if(wid >= B_*S_) return;
  int b = wid >> 11;                       // S_ == 2048
  const int8_t* xr = xq + (size_t)wid*H_;
  const float* ub = u + (size_t)b*H_;
  float acc = 0.f;
#pragma unroll
  for(int j=0;j<8;++j){
    int h = (j*32+lane)*4;
    int w4 = *(const int*)(xr + h);
    acc += ub[h  ]*(float)(int8_t)(w4)
         + ub[h+1]*(float)(int8_t)(w4>>8)
         + ub[h+2]*(float)(int8_t)(w4>>16)
         + ub[h+3]*(float)(int8_t)(w4>>24);
  }
  acc = wred_f(acc);
  if(lane==0) sc[wid] = ((*gk)*ctxr[wid]*acc + qbk[b]) * 0.03125f;   // 1/sqrt(1024)
}

// ---------------- softmax over S, fold c = aw * gamma_v * recip_bs (in place) ----------------
__global__ void softmax_c_kernel(float* __restrict__ sc, const float* __restrict__ ctxr,
                                 const float* __restrict__ gv){
  int b = blockIdx.x;
  float* row = sc + (size_t)b*S_;
  __shared__ float red[256];
  __shared__ float Ms, Ss;
  float m = -3.4e38f;
  for(int s=threadIdx.x;s<S_;s+=256) m = fmaxf(m,row[s]);
  red[threadIdx.x]=m; __syncthreads();
  for(int t=128;t;t>>=1){ if(threadIdx.x<t) red[threadIdx.x]=fmaxf(red[threadIdx.x],red[threadIdx.x+t]); __syncthreads(); }
  if(threadIdx.x==0) Ms = red[0];
  __syncthreads();
  float M = Ms, sum = 0.f;
  for(int s=threadIdx.x;s<S_;s+=256) sum += expf(row[s]-M);
  __syncthreads();
  red[threadIdx.x]=sum; __syncthreads();
  for(int t=128;t;t>>=1){ if(threadIdx.x<t) red[threadIdx.x]+=red[threadIdx.x+t]; __syncthreads(); }
  if(threadIdx.x==0) Ss = red[0];
  __syncthreads();
  float inv = 1.f/Ss, g = *gv;
  for(int s=threadIdx.x;s<S_;s+=256)
    row[s] = expf(row[s]-M)*inv * g * ctxr[(size_t)b*S_+s];
}

// ---------------- y partials: ypart[b,ch,h] = sum_{s in chunk} c[b,s]*xq[b,s,h] ----------------
__global__ void yacc_kernel(const int8_t* __restrict__ xq, const float* __restrict__ c,
                            float* __restrict__ ypart){
  int b = blockIdx.x, ch = blockIdx.y;
  int h0 = threadIdx.x*4;
  float a0=0.f,a1=0.f,a2=0.f,a3=0.f;
  const int8_t* base = xq + ((size_t)b*S_ + (size_t)ch*128)*H_;
  const float*  cb   = c  +  (size_t)b*S_ + (size_t)ch*128;
  for(int s=0;s<128;++s){
    float co = cb[s];
    int w4 = *(const int*)(base + (size_t)s*H_ + h0);
    a0 += co*(float)(int8_t)(w4);
    a1 += co*(float)(int8_t)(w4>>8);
    a2 += co*(float)(int8_t)(w4>>16);
    a3 += co*(float)(int8_t)(w4>>24);
  }
  float* yp = ypart + ((size_t)(b*16+ch))*H_ + h0;
  yp[0]=a0; yp[1]=a1; yp[2]=a2; yp[3]=a3;
}

__global__ void yred_kernel(const float* __restrict__ ypart, float* __restrict__ y){
  int t = blockIdx.x*256 + threadIdx.x;          // over B*H
  int b = t >> 10, h = t & 1023;
  float s = 0.f;
  for(int ch=0;ch<16;++ch) s += ypart[((size_t)(b*16+ch))*H_ + h];
  y[t] = s;
}

// ---------------- cv[b,p] = Wv_int[p,:] . y[b,:] + bv[p]  (gamma_v folded into y) ----------------
__global__ void cvy_kernel(const int8_t* __restrict__ Wvq, const float* __restrict__ y,
                           const float* __restrict__ bv, float* __restrict__ cv){
  int gtid = blockIdx.x*blockDim.x + threadIdx.x;
  int wid = gtid >> 5, lane = gtid & 31;
  if(wid >= B_*P_) return;
  int b = wid >> 10, p = wid & 1023;
  const int8_t* wr = Wvq + (size_t)p*H_;
  const float*  yb = y   + (size_t)b*H_;
  float acc = 0.f;
#pragma unroll
  for(int j=0;j<8;++j){
    int h = (j*32+lane)*4;
    int w4 = *(const int*)(wr + h);
    acc += yb[h  ]*(float)(int8_t)(w4)
         + yb[h+1]*(float)(int8_t)(w4>>8)
         + yb[h+2]*(float)(int8_t)(w4>>16)
         + yb[h+3]*(float)(int8_t)(w4>>24);
  }
  acc = wred_f(acc);
  if(lane==0) cv[(size_t)b*P_ + p] = acc + bv[p];
}

__global__ void cat_kernel(const float* __restrict__ plan, const float* __restrict__ cv,
                           float* __restrict__ cat){
  int t = blockIdx.x*256 + threadIdx.x;          // over B*2P
  if(t >= B_*2*P_) return;
  int b = t >> 11, c = t & 2047;
  cat[t] = (c < P_) ? plan[(size_t)b*P_ + c] : cv[(size_t)b*P_ + c - P_];
}

// ---------------- GRU with h=0: plan = (1-sig(iz+hz)) * tanh(inn + sig(ir+hr)*hn), gh = bhh ----
__global__ void gru_kernel(const float* __restrict__ gi, const float* __restrict__ bhh,
                           float* __restrict__ plan, float* __restrict__ st){
  int t = blockIdx.x*256 + threadIdx.x;          // over B*P
  if(t >= B_*P_) return;
  int b = t >> 10, p = t & 1023;
  const float* g = gi + (size_t)b*3*P_;
  float r = 1.f/(1.f+expf(-(g[p]        + bhh[p])));
  float z = 1.f/(1.f+expf(-(g[P_  + p]  + bhh[P_  + p])));
  float n = tanhf(      g[2*P_+ p]  + r * bhh[2*P_+ p]);
  float v = (1.f - z)*n;
  plan[t] = v;
  st[t]   = v;
}

// ---------------- gate[b,i] = sigmoid( int8dot(planq_b, gateq_i)*gamma_i*recip_b + gb_i ) --------
__global__ void gate_kernel(const int8_t* __restrict__ planq, const float* __restrict__ precip,
                            const int8_t* __restrict__ gq, const float* __restrict__ ggamma,
                            const float* __restrict__ gb, float* __restrict__ gates, int step){
  int gtid = blockIdx.x*blockDim.x + threadIdx.x;
  int wid = gtid >> 5, lane = gtid & 31;
  if(wid >= B_) return;
  const int8_t* xr = planq + (size_t)wid*P_;
  int acc = 0;
#pragma unroll
  for(int j=0;j<8;++j){
    int h = (j*32+lane)*4;
    int x4 = *(const int*)(xr + h);
    int w4 = *(const int*)(gq + h);
    acc += (int)(int8_t)(x4)     * (int)(int8_t)(w4)
         + (int)(int8_t)(x4>>8)  * (int)(int8_t)(w4>>8)
         + (int)(int8_t)(x4>>16) * (int)(int8_t)(w4>>16)
         + (int)(int8_t)(x4>>24) * (int)(int8_t)(w4>>24);
  }
  acc = wred_i(acc);
  if(lane==0){
    float v = (*ggamma)*precip[wid]*(float)acc + *gb;
    gates[wid*ST_ + step] = 1.f/(1.f+expf(-v));
  }
}

__global__ void imp_kernel(const float* __restrict__ gates, float* __restrict__ out){
  int b = threadIdx.x;
  if(b >= B_) return;
  float g[ST_], m = -3.4e38f;
  for(int i=0;i<ST_;++i){ g[i]=gates[b*ST_+i]; m=fmaxf(m,g[i]); }
  float s = 0.f;
  for(int i=0;i<ST_;++i){ g[i]=expf(g[i]-m); s+=g[i]; }
  float inv = 1.f/s;
  for(int i=0;i<ST_;++i) out[b*ST_+i] = g[i]*inv;
}

__global__ void flat_kernel(const float* __restrict__ st, float* __restrict__ flat){
  int t = blockIdx.x*256 + threadIdx.x;          // over B*5P
  if(t >= B_*ST_*P_) return;
  int b = t / (ST_*P_);
  int rem = t - b*(ST_*P_);
  int i = rem >> 10, p = rem & 1023;
  flat[t] = st[((size_t)i*B_ + b)*P_ + p];
}

// =====================================================================================
extern "C" void kernel_launch(void* const* d_in, const int* in_sizes, int n_in,
                              void* d_out, int out_size, void* d_ws, size_t ws_size,
                              hipStream_t stream){
  (void)in_sizes; (void)n_in; (void)out_size; (void)ws_size;
  const float* ctx_state = (const float*)d_in[0];
  const float* ctx_mem   = (const float*)d_in[1];
  const float* ce_w  = (const float*)d_in[2];  const float* ce_b  = (const float*)d_in[3];
  const float* wih   = (const float*)d_in[4];  const float* bih   = (const float*)d_in[5];
  /* d_in[6] gru_whh: dead — multiplied by bitlinear-quantized zeros */
  const float* bhh   = (const float*)d_in[7];
  const float* a_wq  = (const float*)d_in[8];  const float* a_bq  = (const float*)d_in[9];
  const float* a_wk  = (const float*)d_in[10]; const float* a_bk  = (const float*)d_in[11];
  const float* a_wv  = (const float*)d_in[12]; const float* a_bv  = (const float*)d_in[13];
  const float* a_wc  = (const float*)d_in[14]; const float* a_bc  = (const float*)d_in[15];
  const float* agg_w = (const float*)d_in[16]; const float* agg_b = (const float*)d_in[17];
  const float* out_w = (const float*)d_in[18]; const float* out_b = (const float*)d_in[19];
  const float* gt_w  = (const float*)d_in[20]; const float* gt_b  = (const float*)d_in[21];

  float* OUT        = (float*)d_out;
  float* out_emb    = OUT;                        // [B,H]
  float* out_states = OUT + B_*H_;                // [ST,B,P]
  float* out_imp    = out_states + ST_*B_*P_;     // [B,ST]

  // ---- workspace layout ----
  char* ws = (char*)d_ws;
  size_t off = 0;
  auto alloc = [&](size_t bytes){ size_t o = off; off = (off + bytes + 255) & ~(size_t)255; return o; };
  int8_t* xq    = (int8_t*)(ws + alloc((size_t)B_*S_*H_));
  float*  ctxr  = (float*) (ws + alloc((size_t)B_*S_*4));
  int8_t* qce   = (int8_t*)(ws + alloc((size_t)P_*H_));
  int8_t* qih   = (int8_t*)(ws + alloc((size_t)3*P_*P_));
  int8_t* qwq   = (int8_t*)(ws + alloc((size_t)A_*P_));
  int8_t* qwk   = (int8_t*)(ws + alloc((size_t)A_*H_));
  int8_t* qwv   = (int8_t*)(ws + alloc((size_t)P_*H_));
  int8_t* qwc   = (int8_t*)(ws + alloc((size_t)P_*2*P_));
  int8_t* qagg  = (int8_t*)(ws + alloc((size_t)P_*5*P_));
  int8_t* qout  = (int8_t*)(ws + alloc((size_t)H_*P_));
  int8_t* qgate = (int8_t*)(ws + alloc((size_t)ST_*P_));
  float*  parts = (float*) (ws + alloc(2048*4));
  float*  gam   = (float*) (ws + alloc(64*4));
  float*  plan  = (float*) (ws + alloc((size_t)B_*P_*4));
  float*  qv    = (float*) (ws + alloc((size_t)B_*A_*4));
  float*  u     = (float*) (ws + alloc((size_t)B_*H_*4));
  float*  qbk   = (float*) (ws + alloc(B_*4));
  float*  sc    = (float*) (ws + alloc((size_t)B_*S_*4));
  float*  ypart = (float*) (ws + alloc((size_t)B_*16*H_*4));
  float*  y     = (float*) (ws + alloc((size_t)B_*H_*4));
  float*  cv    = (float*) (ws + alloc((size_t)B_*P_*4));
  float*  cat   = (float*) (ws + alloc((size_t)B_*2*P_*4));
  float*  gi    = (float*) (ws + alloc((size_t)B_*3*P_*4));
  float*  gates = (float*) (ws + alloc((size_t)B_*ST_*4));
  float*  flat  = (float*) (ws + alloc((size_t)B_*5*P_*4));
  float*  aggr  = (float*) (ws + alloc((size_t)B_*P_*4));
  int8_t* actq  = (int8_t*)(ws + alloc((size_t)B_*5*P_));
  float*  actr  = (float*) (ws + alloc(64*4));

  enum { ICE=0, IWIH, IWQ, IWK, IWV, IWC, IAGG, IOUT, IG0 };

  auto wquant = [&](const float* w, int n, int8_t* dst, int idx){
    int nb = (n + 255) / 256; if(nb > 1024) nb = 1024;
    absum_kernel<<<nb, 256, 0, stream>>>(w, n, parts);
    fin_gamma_kernel<<<1, 256, 0, stream>>>(parts, nb, 1.f/(float)n, gam + idx);
    quantw_kernel<<<nb, 256, 0, stream>>>(w, n, gam + idx, dst);
  };
  auto gemm = [&](const int8_t* Xq, const float* xr, const int8_t* Wq, const float* g,
                  const float* bias, const float* addsrc, float* out, int N, int K, int act){
    gemm_iu8_kernel<<<dim3(N/64, B_/16), 128, 0, stream>>>(Xq, xr, Wq, g, bias, addsrc, out, N, K, act);
  };
  auto qact = [&](const float* x, int rows, int cols){
    quant_rows_kernel<<<rows, 256, 0, stream>>>(x, cols, actq, actr);
  };

  // ---- quantize all weights (ternary int8 + gamma) ----
  wquant(ce_w,  P_*H_,   qce,  ICE);
  wquant(wih,   3*P_*P_, qih,  IWIH);
  wquant(a_wq,  A_*P_,   qwq,  IWQ);
  wquant(a_wk,  A_*H_,   qwk,  IWK);
  wquant(a_wv,  P_*H_,   qwv,  IWV);
  wquant(a_wc,  P_*2*P_, qwc,  IWC);
  wquant(agg_w, P_*5*P_, qagg, IAGG);
  wquant(out_w, H_*P_,   qout, IOUT);
  for(int i=0;i<ST_;++i) wquant(gt_w + i*P_, P_, qgate + i*P_, IG0 + i);

  // ---- quantize context_memory once (step-invariant): 256MB fp32 -> 64MB int8 ----
  quant_rows_kernel<<<B_*S_, 256, 0, stream>>>(ctx_mem, H_, xq, ctxr);

  // ---- plan = tanh(bitlinear(context_state, ce)) ----
  qact(ctx_state, B_, H_);
  gemm(actq, actr, qce, gam+ICE, ce_b, nullptr, plan, P_, H_, 1);
  qact(plan, B_, P_);

  for(int i=0;i<ST_;++i){
    // q = bitlinear(plan, att_wq)
    gemm(actq, actr, qwq, gam+IWQ, a_bq, nullptr, qv, A_, P_, 0);
    // scores = (gamma_k/s) * (Wk^T q).xq / sqrt(A)  (+ q.bk/sqrt(A))
    qbk_kernel<<<B_, 256, 0, stream>>>(qv, a_bk, qbk);
    u_kernel<<<B_, 256, 0, stream>>>(qv, qwk, u);
    scores_kernel<<<(B_*S_*32)/256, 256, 0, stream>>>(xq, ctxr, u, qbk, gam+IWK, sc);
    softmax_c_kernel<<<B_, 256, 0, stream>>>(sc, ctxr, gam+IWV);
    // cv = Wv . (sum_s aw*gamma_v/s * xq_s) + bv
    yacc_kernel<<<dim3(B_,16), 256, 0, stream>>>(xq, sc, ypart);
    yred_kernel<<<(B_*H_)/256, 256, 0, stream>>>(ypart, y);
    cvy_kernel<<<(B_*P_*32)/256, 256, 0, stream>>>(qwv, y, a_bv, cv);
    // plan += bitlinear(concat[plan,cv], att_wc)
    cat_kernel<<<(B_*2*P_)/256, 256, 0, stream>>>(plan, cv, cat);
    qact(cat, B_, 2*P_);
    gemm(actq, actr, qwc, gam+IWC, a_bc, plan, plan, P_, 2*P_, 0);
    // GRU (hidden=0, gh = bhh)
    qact(plan, B_, P_);
    gemm(actq, actr, qih, gam+IWIH, bih, nullptr, gi, 3*P_, P_, 0);
    gru_kernel<<<(B_*P_)/256, 256, 0, stream>>>(gi, bhh, plan, out_states + (size_t)i*B_*P_);
    // gate
    qact(plan, B_, P_);
    gate_kernel<<<(B_*32)/256, 256, 0, stream>>>(actq, actr, qgate + i*P_, gam+IG0+i, gt_b + i, gates, i);
  }

  imp_kernel<<<1, 32, 0, stream>>>(gates, out_imp);

  // aggregated = bitlinear(flat, agg_w); plan_embedding = bitlinear(aggregated, out_w)
  flat_kernel<<<(B_*ST_*P_)/256, 256, 0, stream>>>(out_states, flat);
  qact(flat, B_, 5*P_);
  gemm(actq, actr, qagg, gam+IAGG, agg_b, nullptr, aggr, P_, 5*P_, 0);
  qact(aggr, B_, P_);
  gemm(actq, actr, qout, gam+IOUT, out_b, nullptr, out_emb, H_, P_, 0);
}
